// DualEncoderEpsNetwork_60232621359485
// MI455X (gfx1250) — matrix operand order repaired and checked by reading, not compiled, not run on hardware
//
#include <hip/hip_runtime.h>
#include <hip/hip_bf16.h>
#include <cstdint>
#include <cstddef>

// ---------------------------------------------------------------------------
// Types / constants
// ---------------------------------------------------------------------------
typedef __bf16 bf16_t;
typedef bf16_t v8bf  __attribute__((ext_vector_type(8)));
typedef bf16_t v16bf __attribute__((ext_vector_type(16)));
typedef float  v8f   __attribute__((ext_vector_type(8)));

#define HDIM 256      // hidden dim
#define APM  25       // atoms per molecule
#define EPM  600      // edges per molecule (25*24)
#define NLAYER 4

__device__ __forceinline__ float siluf(float x) { return x / (1.0f + __expf(-x)); }
__device__ __forceinline__ float sspf(float x) {
  float sp = (x > 15.0f) ? x : log1pf(__expf(x));
  return sp - 0.6931472f;
}

// ---------------------------------------------------------------------------
// WMMA fragment loaders (bf16 16x16x32).
// A fragment: lane l<16 holds row M=l, K={0..7,16..23}; lane l+16 holds K={8..15,24..31}
// B fragment: lane l<16 holds col N=l, K={0..7,16..23}; lane l+16 holds K={8..15,24..31}
// Bt is the transposed weight: Bt[n][k] row-major -> contiguous K per lane.
// ---------------------------------------------------------------------------
__device__ __forceinline__ v16bf load_fragA(const bf16_t* p) {
  v8bf lo = *reinterpret_cast<const v8bf*>(p);
  v8bf hi = *reinterpret_cast<const v8bf*>(p + 16);
  v16bf r;
#pragma unroll
  for (int i = 0; i < 8; ++i) { r[i] = lo[i]; r[i + 8] = hi[i]; }
  return r;
}
__device__ __forceinline__ v16bf load_fragB(const bf16_t* p) {
  v8bf lo = *reinterpret_cast<const v8bf*>(p);
  v8bf hi = *reinterpret_cast<const v8bf*>(p + 16);
  v16bf r;
#pragma unroll
  for (int i = 0; i < 8; ++i) { r[i] = lo[i]; r[i + 8] = hi[i]; }
  return r;
}

// ---------------------------------------------------------------------------
// Fused epilogue.
// ACT:  0 = none, 1 = silu, 2 = shifted-softplus
// MODE: 0 = f32 out, 1 = bf16 out, 3 = bond-gate + concat (mlp_d path),
//       4 = residual h += ssp(acc+bias), writes f32 h and bf16 h
// ---------------------------------------------------------------------------
template<int ACT, int MODE>
__device__ __forceinline__ void epi_tile(
    v8f acc, int n, int mBase, int M, int N,
    const float* __restrict__ bias,
    float* __restrict__ outF, bf16_t* __restrict__ outB,
    const int* __restrict__ tr, const int* __restrict__ tp,
    const float* __restrict__ bemb, const float* __restrict__ hin)
{
  float b = bias ? bias[n] : 0.0f;
#pragma unroll
  for (int i = 0; i < 8; ++i) {
    int m = mBase + i;
    if (m >= M) continue;
    float v = acc[i] + b;
    if (MODE == 4) {
      float r = hin[(size_t)m * N + n] + sspf(v);
      outF[(size_t)m * N + n] = r;
      outB[(size_t)m * N + n] = (bf16_t)r;
    } else if (MODE == 3) {
      float rr = v * bemb[tr[m] * HDIM + n];
      float pp = v * bemb[tp[m] * HDIM + n];
      outB[(size_t)m * (2 * HDIM) + n]        = (bf16_t)rr;
      outB[(size_t)m * (2 * HDIM) + HDIM + n] = (bf16_t)pp;
    } else {
      if (ACT == 1)      v = siluf(v);
      else if (ACT == 2) v = sspf(v);
      if (MODE == 0) outF[(size_t)m * N + n] = v;
      else           outB[(size_t)m * N + n] = (bf16_t)v;
    }
  }
}

// ---------------------------------------------------------------------------
// Tiled bf16 WMMA GEMM:  D[M,N] = epilogue(A[M,K] @ Bt[N,K]^T + bias)
// Block = 256 threads = 8 wave32. Each wave computes a 64x64 tile:
// 4 A-frags x 4 B-frags -> 16 WMMAs per K-step for 16 b128 loads
// (1.0 load/WMMA; register-level reuse instead of LDS round-trips).
// N must be a multiple of 64. Block tile: 128x256 (N=256) or 256x128 (N=128).
// ---------------------------------------------------------------------------
template<int ACT, int MODE>
__global__ __launch_bounds__(256) void gemm_bf16_wmma(
    const bf16_t* __restrict__ A, const bf16_t* __restrict__ Bt,
    const float* __restrict__ bias, int M, int K, int N,
    float* __restrict__ outF, bf16_t* __restrict__ outB,
    const int* __restrict__ tr, const int* __restrict__ tp,
    const float* __restrict__ bemb, const float* __restrict__ hin)
{
  const int wave = threadIdx.x >> 5;   // wave32
  const int lane = threadIdx.x & 31;
  const int colTiles = N >> 6;                        // 64-wide col groups (4 or 2)
  const int colBase  = (wave % colTiles) << 6;
  const int rowsPerBlock = (8 / colTiles) << 6;       // 128 (N=256) or 256 (N=128)
  const int rowBase = blockIdx.x * rowsPerBlock + (wave / colTiles) * 64;
  const int l15 = lane & 15;
  const int hiH = lane >> 4;

  const bf16_t* aPtr = A  + (size_t)(rowBase + l15) * K + (hiH << 3);
  const bf16_t* bPtr = Bt + (size_t)(colBase + l15) * K + (hiH << 3);
  const size_t s16K = (size_t)16 * K;

  v8f acc[4][4];
#pragma unroll
  for (int r = 0; r < 4; ++r)
#pragma unroll
    for (int t = 0; t < 4; ++t)
      acc[r][t] = (v8f){0.f,0.f,0.f,0.f,0.f,0.f,0.f,0.f};

  for (int k = 0; k < K; k += 32) {
    v16bf a[4];
#pragma unroll
    for (int r = 0; r < 4; ++r) a[r] = load_fragA(aPtr + (size_t)r * s16K + k);
#pragma unroll
    for (int t = 0; t < 4; ++t) {
      v16bf bfr = load_fragB(bPtr + (size_t)t * s16K + k);
#pragma unroll
      for (int r = 0; r < 4; ++r)
        acc[r][t] = __builtin_amdgcn_wmma_f32_16x16x32_bf16(
            false, a[r], false, bfr, (short)0, acc[r][t], false, false);
    }
  }

  // D layout: lane<16 -> M 0..7, lane>=16 -> M 8..15 within each 16x16 tile
#pragma unroll
  for (int r = 0; r < 4; ++r) {
    const int mBase = rowBase + 16 * r + (hiH << 3);
#pragma unroll
    for (int t = 0; t < 4; ++t)
      epi_tile<ACT, MODE>(acc[r][t], colBase + 16 * t + l15, mBase, M, N,
                          bias, outF, outB, tr, tp, bemb, hin);
  }
}

// ---------------------------------------------------------------------------
// Weight pre-pack: f32 [K,N] row-major -> bf16 [N,K] row-major (Bt layout)
// ---------------------------------------------------------------------------
__global__ void pack_wT(const float* __restrict__ in, bf16_t* __restrict__ out,
                        int K, int N) {
  int tid = blockIdx.x * 256 + threadIdx.x;
  if (tid >= K * N) return;
  int k = tid / N, n = tid - k * N;
  out[(size_t)n * K + k] = (bf16_t)in[tid];
}

// z = cat(atom_emb[type] + r_feat@afW, (p_feat - r_feat)@afW)  -> h f32 + bf16
__global__ void node_embed(const float* __restrict__ rf, const float* __restrict__ pf,
                           const float* __restrict__ emb, const float* __restrict__ afW,
                           const int* __restrict__ atype,
                           float* __restrict__ h, bf16_t* __restrict__ hb,
                           int Nreal, int Npad) {
  int tid = blockIdx.x * 256 + threadIdx.x;
  if (tid >= Npad * HDIM) return;
  int n = tid >> 8, c = tid & 255;
  float v = 0.0f;
  if (n < Nreal) {
    int cc = c & 127;
    const float* r = rf + (size_t)n * 27;
    float fr = 0.0f;
#pragma unroll
    for (int k = 0; k < 27; ++k) fr += r[k] * afW[k * 128 + cc];
    if (c < 128) {
      v = emb[(size_t)atype[n] * 128 + c] + fr;
    } else {
      const float* p = pf + (size_t)n * 27;
      float fp = 0.0f;
#pragma unroll
      for (int k = 0; k < 27; ++k) fp += p[k] * afW[k * 128 + cc];
      v = fp - fr;
    }
  }
  h[tid]  = v;
  hb[tid] = (bf16_t)v;
}

// Edge distances; also emit d and edge_index sections of d_out.
__global__ void edge_geom(const float* __restrict__ pos, const int* __restrict__ ei,
                          float* __restrict__ dbuf, float* __restrict__ dOut,
                          int* __restrict__ idxOut, int E) {
  int e = blockIdx.x * 256 + threadIdx.x;
  if (e >= E) return;
  int s = ei[e], t = ei[E + e];
  float dx = pos[t * 3 + 0] - pos[s * 3 + 0];
  float dy = pos[t * 3 + 1] - pos[s * 3 + 1];
  float dz = pos[t * 3 + 2] - pos[s * 3 + 2];
  float d = sqrtf(dx * dx + dy * dy + dz * dz);
  dbuf[e] = d;
  dOut[e] = d;
  idxOut[e] = s;
  idxOut[E + e] = t;
}

// t0 = silu(d * ee_W1 + ee_b1)  (rank-1 expansion) -> bf16
__global__ void edge_rbf(const float* __restrict__ dbuf, const float* __restrict__ w1,
                         const float* __restrict__ b1, bf16_t* __restrict__ out, int E) {
  int tid = blockIdx.x * 256 + threadIdx.x;
  if (tid >= E * HDIM) return;
  int e = tid >> 8, c = tid & 255;
  out[tid] = (bf16_t)siluf(dbuf[e] * w1[c] + b1[c]);
}

// Deterministic segment-sum as a gather: node n=g*25+j receives from the 24
// edges e = g*600 + i*24 + (j - (j>i)) with src = g*25+i.
__global__ void cfconv_agg(const float* __restrict__ hl, const bf16_t* __restrict__ Wf,
                           bf16_t* __restrict__ aggb, int Nreal, int Npad) {
  int tid = blockIdx.x * 256 + threadIdx.x;
  if (tid >= Npad * HDIM) return;
  int n = tid >> 8, c = tid & 255;
  float s = 0.0f;
  if (n < Nreal) {
    int g = n / APM, j = n - g * APM;
    int nb = g * APM, eb = g * EPM;
    for (int i = 0; i < APM; ++i) {
      if (i == j) continue;
      int jp = (j > i) ? (j - 1) : j;
      int e = eb + i * 24 + jp;
      s += hl[(size_t)(nb + i) * HDIM + c] * (float)Wf[(size_t)e * HDIM + c];
    }
  }
  aggb[tid] = (bf16_t)s;
}

// h_pair = cat(h[src]*h[dst], edge_attr) -> bf16 [E, 512]
__global__ void pair_feat(const float* __restrict__ h, const bf16_t* __restrict__ eattr,
                          const int* __restrict__ ei, bf16_t* __restrict__ out, int E) {
  int tid = blockIdx.x * 256 + threadIdx.x;
  if (tid >= E * HDIM) return;
  int e = tid >> 8, c = tid & 255;
  int s = ei[e], t = ei[E + e];
  out[(size_t)e * (2 * HDIM) + c] =
      (bf16_t)(h[(size_t)s * HDIM + c] * h[(size_t)t * HDIM + c]);
  out[(size_t)e * (2 * HDIM) + HDIM + c] = eattr[tid];
}

// edge_inv = t2 @ g_W3 + g_b3   ([E,128] x [128,1])
__global__ void final_dot(const bf16_t* __restrict__ t2, const float* __restrict__ w3,
                          const float* __restrict__ b3, float* __restrict__ out, int E) {
  int e = blockIdx.x * 256 + threadIdx.x;
  if (e >= E) return;
  const bf16_t* r = t2 + (size_t)e * 128;
  float s = b3[0];
#pragma unroll 8
  for (int c = 0; c < 128; ++c) s += (float)r[c] * w3[c];
  out[e] = s;
}

// ---------------------------------------------------------------------------
// Host-side orchestration
// ---------------------------------------------------------------------------
extern "C" void kernel_launch(void* const* d_in, const int* in_sizes, int n_in,
                              void* d_out, int out_size, void* d_ws, size_t ws_size,
                              hipStream_t stream) {
  (void)n_in; (void)out_size; (void)ws_size;
  const float* pos   = (const float*)d_in[0];
  const float* rfeat = (const float*)d_in[1];
  const float* pfeat = (const float*)d_in[2];
  const float* emb   = (const float*)d_in[3];
  const float* afW   = (const float*)d_in[4];
  const float* bemb  = (const float*)d_in[5];
  const float* eeW1  = (const float*)d_in[6];
  const float* eeB1  = (const float*)d_in[7];
  const float* eeW2  = (const float*)d_in[8];
  const float* eeB2  = (const float*)d_in[9];
  const float* ecW1  = (const float*)d_in[10];
  const float* ecB1  = (const float*)d_in[11];
  const float* ecW2  = (const float*)d_in[12];
  const float* ecB2  = (const float*)d_in[13];
  const float* fW1   = (const float*)d_in[14];
  const float* fB1   = (const float*)d_in[15];
  const float* fW2   = (const float*)d_in[16];
  const float* fB2   = (const float*)d_in[17];
  const float* lin1  = (const float*)d_in[18];
  const float* lin2  = (const float*)d_in[19];
  const float* lin2b = (const float*)d_in[20];
  const float* gW1   = (const float*)d_in[21];
  const float* gB1   = (const float*)d_in[22];
  const float* gW2   = (const float*)d_in[23];
  const float* gB2   = (const float*)d_in[24];
  const float* gW3   = (const float*)d_in[25];
  const float* gB3   = (const float*)d_in[26];
  const int*   atype = (const int*)d_in[27];
  const int*   ei    = (const int*)d_in[28];
  const int*   etr   = (const int*)d_in[29];
  const int*   etp   = (const int*)d_in[30];

  const int N    = in_sizes[0] / 3;           // 5000 nodes
  const int E    = in_sizes[29];              // 120000 edges
  const int Npad = (N + 127) & ~127;          // node GEMMs: 128-row blocks
  const int Epad = (E + 255) & ~255;          // edge GEMMs: up to 256-row blocks
  const int Hs   = HDIM;

  // ---- carve workspace ----
  char* base = (char*)d_ws;
  size_t off = 0;
  auto carve = [&](size_t bytes) -> char* {
    char* p = base + off;
    off = (off + bytes + 255) & ~(size_t)255;
    return p;
  };
  bf16_t* wEE2 = (bf16_t*)carve((size_t)Hs * Hs * 2);
  bf16_t* wEC1 = (bf16_t*)carve((size_t)Hs * 2 * Hs * 2);
  bf16_t* wEC2 = (bf16_t*)carve((size_t)Hs * Hs * 2);
  bf16_t* wF1  = (bf16_t*)carve((size_t)NLAYER * Hs * Hs * 2);
  bf16_t* wF2  = (bf16_t*)carve((size_t)NLAYER * Hs * Hs * 2);
  bf16_t* wL1  = (bf16_t*)carve((size_t)NLAYER * Hs * Hs * 2);
  bf16_t* wL2  = (bf16_t*)carve((size_t)NLAYER * Hs * Hs * 2);
  bf16_t* wG1  = (bf16_t*)carve((size_t)Hs * 2 * Hs * 2);
  bf16_t* wG2  = (bf16_t*)carve((size_t)128 * Hs * 2);
  bf16_t* ebuf0 = (bf16_t*)carve((size_t)Epad * 2 * Hs * 2);  // [E,512] cat / h_pair
  bf16_t* ebuf2 = (bf16_t*)carve((size_t)Epad * Hs * 2);      // bf16 scratch
  bf16_t* ebuf3 = (bf16_t*)carve((size_t)Epad * Hs * 2);      // bf16 scratch (Wf, t2)
  bf16_t* eattr = (bf16_t*)carve((size_t)Epad * Hs * 2);      // persistent edge_attr
  float*  dbuf  = (float*)carve((size_t)E * 4);
  float*  hF    = (float*)carve((size_t)Npad * Hs * 4);
  bf16_t* hB    = (bf16_t*)carve((size_t)Npad * Hs * 2);
  float*  hlF   = (float*)carve((size_t)Npad * Hs * 4);
  bf16_t* aggB  = (bf16_t*)carve((size_t)Npad * Hs * 2);

  float* outInv = (float*)d_out;            // [E]
  int*   idxOut = ((int*)d_out) + E;        // [2,E] (raw int bits)
  float* dOut   = ((float*)d_out) + 3 * E;  // [E]

  auto cdiv = [](int a, int b) { return (a + b - 1) / b; };
  auto gblk = [&](int M, int Nc) { int rpb = (8 / (Nc >> 6)) << 6; return cdiv(M, rpb); };

  // ---- pack transposed bf16 weights (cheap one-time per launch) ----
  auto packN = [&](const float* in, bf16_t* out, int K, int Nc) {
    int tot = K * Nc;
    pack_wT<<<cdiv(tot, 256), 256, 0, stream>>>(in, out, K, Nc);
  };
  packN(eeW2, wEE2, Hs, Hs);
  packN(ecW1, wEC1, 2 * Hs, Hs);
  packN(ecW2, wEC2, Hs, Hs);
  for (int l = 0; l < NLAYER; ++l) {
    size_t o = (size_t)l * Hs * Hs;
    packN(fW1 + o,  wF1 + o, Hs, Hs);
    packN(fW2 + o,  wF2 + o, Hs, Hs);
    packN(lin1 + o, wL1 + o, Hs, Hs);
    packN(lin2 + o, wL2 + o, Hs, Hs);
  }
  packN(gW1, wG1, 2 * Hs, Hs);
  packN(gW2, wG2, Hs, 128);

  // ---- node embedding + edge geometry ----
  node_embed<<<cdiv(Npad * Hs, 256), 256, 0, stream>>>(rfeat, pfeat, emb, afW, atype,
                                                       hF, hB, N, Npad);
  edge_geom<<<cdiv(E, 256), 256, 0, stream>>>(pos, ei, dbuf, dOut, idxOut, E);
  edge_rbf<<<cdiv(E * Hs, 256), 256, 0, stream>>>(dbuf, eeW1, eeB1, ebuf2, E);

  // ---- mlp_d GEMM + bond-gate-concat epilogue -> ebuf0 [E,512] ----
  gemm_bf16_wmma<0, 3><<<gblk(E, Hs), 256, 0, stream>>>(
      ebuf2, wEE2, eeB2, E, Hs, Hs, nullptr, ebuf0, etr, etp, bemb, nullptr);
  // ---- edge_cat MLP: silu(cat@ec_W1+b) @ ec_W2 + b -> eattr ----
  gemm_bf16_wmma<1, 1><<<gblk(E, Hs), 256, 0, stream>>>(
      ebuf0, wEC1, ecB1, E, 2 * Hs, Hs, nullptr, ebuf2, nullptr, nullptr, nullptr, nullptr);
  gemm_bf16_wmma<0, 1><<<gblk(E, Hs), 256, 0, stream>>>(
      ebuf2, wEC2, ecB2, E, Hs, Hs, nullptr, eattr, nullptr, nullptr, nullptr, nullptr);

  // ---- 4 SchNet CFConv interaction blocks ----
  for (int l = 0; l < NLAYER; ++l) {
    size_t o = (size_t)l * Hs * Hs;
    // edge filter W = ssp(eattr@fW1+b) @ fW2 + b
    gemm_bf16_wmma<2, 1><<<gblk(E, Hs), 256, 0, stream>>>(
        eattr, wF1 + o, fB1 + (size_t)l * Hs, E, Hs, Hs,
        nullptr, ebuf2, nullptr, nullptr, nullptr, nullptr);
    gemm_bf16_wmma<0, 1><<<gblk(E, Hs), 256, 0, stream>>>(
        ebuf2, wF2 + o, fB2 + (size_t)l * Hs, E, Hs, Hs,
        nullptr, ebuf3, nullptr, nullptr, nullptr, nullptr);
    // hl = h @ lin1 (no bias)
    gemm_bf16_wmma<0, 0><<<gblk(Npad, Hs), 256, 0, stream>>>(
        hB, wL1 + o, nullptr, Npad, Hs, Hs,
        hlF, nullptr, nullptr, nullptr, nullptr, nullptr);
    // deterministic gather segment-sum: agg = sum_src hl[src] * W[e]
    cfconv_agg<<<cdiv(Npad * Hs, 256), 256, 0, stream>>>(hlF, ebuf3, aggB, N, Npad);
    // h = h + ssp(agg @ lin2 + b)  (fused residual epilogue, in-place)
    gemm_bf16_wmma<0, 4><<<gblk(Npad, Hs), 256, 0, stream>>>(
        aggB, wL2 + o, lin2b + (size_t)l * Hs, Npad, Hs, Hs,
        hF, hB, nullptr, nullptr, nullptr, hF);
  }

  // ---- pair head: cat(h_src*h_dst, edge_attr) -> 2H -> H -> H/2 -> 1 ----
  pair_feat<<<cdiv(E * Hs, 256), 256, 0, stream>>>(hF, eattr, ei, ebuf0, E);
  gemm_bf16_wmma<1, 1><<<gblk(E, Hs), 256, 0, stream>>>(
      ebuf0, wG1, gB1, E, 2 * Hs, Hs, nullptr, ebuf2, nullptr, nullptr, nullptr, nullptr);
  gemm_bf16_wmma<1, 1><<<gblk(E, 128), 256, 0, stream>>>(
      ebuf2, wG2, gB2, E, Hs, 128, nullptr, ebuf3, nullptr, nullptr, nullptr, nullptr);
  final_dot<<<cdiv(E, 256), 256, 0, stream>>>(ebuf3, gW3, gB3, outInv, E);
}
